// PhysicsBiasedAttention_6253472383614
// MI455X (gfx1250) — matrix operand order repaired and checked by reading, not compiled, not run on hardware
//
#include <hip/hip_runtime.h>
#include <math.h>

// ---------------- problem constants ----------------
#define BATCH   2
#define SEQ     4096
#define DMODEL  512
#define NHEADS  8
#define DK      64
#define BL      (BATCH * SEQ)      // 8192
#define BH      (BATCH * NHEADS)   // 16

typedef __bf16 bf16_t;
typedef __attribute__((ext_vector_type(16))) __bf16 v16bf;
typedef __attribute__((ext_vector_type(8)))  __bf16 v8bf;
typedef __attribute__((ext_vector_type(8)))  float  v8f;
typedef __attribute__((ext_vector_type(4)))  int    v4i;

union AFrag { v16bf v; v8bf h[2]; };
union CFrag { v8f   v; float f[8]; };

// ---------------------------------------------------------------------------
// WMMA fragment loaders (CDNA5 16x16x32 bf16 layouts, wave32)
//
// A (16x32, MxK): lane m(0..15)=row m.  elems 0..7  : K = k0 + (lane<16?0:8)+i
//                                       elems 8..15 : K = k0 + 16 + (lane<16?0:8)+i
// B (32x16, KxN): lane n(0..15)=col n.  elems 0..7  : K = k0 + (lane<16?0:8)+i
//                                       elems 8..15 : K = k0 + 16 + (lane<16?0:8)+i
// C (16x16 f32): VGPR v, lanes 0..15 -> (M=v, N=lane); lanes 16..31 -> (M=v+8, N=lane-16)
// ---------------------------------------------------------------------------
__device__ __forceinline__ v16bf load_frag_a(const bf16_t* base, int ld, int k0, int lane) {
    const int m  = lane & 15;
    const int hi = (lane >> 4) & 1;
    const bf16_t* p = base + (size_t)m * ld + k0 + hi * 8;
    AFrag f;
    f.h[0] = *(const v8bf*)(p);
    f.h[1] = *(const v8bf*)(p + 16);
    return f.v;
}

__device__ __forceinline__ v16bf load_frag_b(const bf16_t* base, int ld, int k0, int lane) {
    const int n  = lane & 15;
    const int hi = (lane >> 4) & 1;
    const bf16_t* p = base + (size_t)n * ld + k0 + hi * 8;
    AFrag f;
    f.h[0] = *(const v8bf*)(p);
    f.h[1] = *(const v8bf*)(p + 16);
    return f.v;
}

__device__ __forceinline__ v8f wmma_bf16(v16bf a, v16bf b, v8f c) {
    // (neg_a, A, neg_b, B, c_mod, C, reuse_a, reuse_b)
    return __builtin_amdgcn_wmma_f32_16x16x32_bf16(false, a, false, b, (short)0, c, false, false);
}

// ---------------------------------------------------------------------------
// Async global->LDS staging (CDNA5 GLOBAL_LOAD_ASYNC_TO_LDS_B128, ASYNCcnt).
// Builtin signature (from hipcc diagnostic): (v4i AS1*, v4i AS3*, Imm, Imm).
// Guarded: falls back to a register-bounce copy if the builtin is absent.
// ---------------------------------------------------------------------------
#if defined(__has_builtin)
#if __has_builtin(__builtin_amdgcn_global_load_async_to_lds_b128)
#define PBA_HAVE_ASYNC_LDS 1
#endif
#endif

#ifdef PBA_HAVE_ASYNC_LDS
typedef __attribute__((address_space(1))) v4i as1_v4i;
typedef __attribute__((address_space(3))) v4i as3_v4i;
__device__ __forceinline__ void pba_copy16_to_lds(const bf16_t* g, bf16_t* l) {
    as1_v4i* gp = (as1_v4i*)(unsigned long long)g;
    // generic LDS pointer: low 32 bits are the LDS byte offset (ISA aperture rule)
    as3_v4i* lp = (as3_v4i*)(unsigned)(unsigned long long)l;
    __builtin_amdgcn_global_load_async_to_lds_b128(gp, lp, 0, 0);
}
__device__ __forceinline__ void pba_wait_async() {
    asm volatile("s_wait_asynccnt 0" ::: "memory");
}
#else
__device__ __forceinline__ void pba_copy16_to_lds(const bf16_t* g, bf16_t* l) {
    *(v8bf*)l = *(const v8bf*)g;
}
__device__ __forceinline__ void pba_wait_async() {}
#endif

// ---------------------------------------------------------------------------
// Kernel 1a: fp32 -> bf16 elementwise
// ---------------------------------------------------------------------------
__global__ void pba_convert_x_kernel(const float* __restrict__ x, bf16_t* __restrict__ xb, int n) {
    int i = blockIdx.x * blockDim.x + threadIdx.x;
    if (i < n) xb[i] = (bf16_t)x[i];
}

// ---------------------------------------------------------------------------
// Kernel 1b: W[k][n] fp32 -> Wt[n][k] bf16  (all four weights via blockIdx.z)
// ---------------------------------------------------------------------------
__global__ void pba_transpose_w_kernel(const float* __restrict__ w0, const float* __restrict__ w1,
                                       const float* __restrict__ w2, const float* __restrict__ w3,
                                       bf16_t* __restrict__ t0, bf16_t* __restrict__ t1,
                                       bf16_t* __restrict__ t2, bf16_t* __restrict__ t3) {
    const float* src = (blockIdx.z == 0) ? w0 : (blockIdx.z == 1) ? w1 : (blockIdx.z == 2) ? w2 : w3;
    bf16_t*      dst = (blockIdx.z == 0) ? t0 : (blockIdx.z == 1) ? t1 : (blockIdx.z == 2) ? t2 : t3;
    int i = blockIdx.x * blockDim.x + threadIdx.x;     // over DMODEL*DMODEL
    int k = i >> 9, n = i & (DMODEL - 1);
    dst[(size_t)n * DMODEL + k] = (bf16_t)src[i];
}

// ---------------------------------------------------------------------------
// Kernel 2: QKV projection GEMM (M=BL, N=DMODEL, K=DMODEL), bf16 WMMA.
//   blockIdx.z selects Q/K/V.  Each wave -> 64x64 tile (16 C frags).
//   Q -> [bh][l][dk], K -> [bh][l][dk], V -> transposed [bh][d][L]
// ---------------------------------------------------------------------------
__global__ void __launch_bounds__(128) pba_qkv_gemm_kernel(
    const bf16_t* __restrict__ xb,
    const bf16_t* __restrict__ wtq, const bf16_t* __restrict__ wtk, const bf16_t* __restrict__ wtv,
    const float* __restrict__ bq, const float* __restrict__ bk, const float* __restrict__ bv,
    bf16_t* __restrict__ Qh, bf16_t* __restrict__ Kh, bf16_t* __restrict__ Vt)
{
    const int lane  = threadIdx.x & 31;
    const int wave  = threadIdx.x >> 5;
    const int which = blockIdx.z;
    const bf16_t* wt  = (which == 0) ? wtq : (which == 1) ? wtk : wtv;
    const float* bias = (which == 0) ? bq  : (which == 1) ? bk  : bv;

    const int tile = blockIdx.x * 4 + wave;     // 128 * 8 = 1024 tiles
    const int m0 = (tile >> 3) * 64;
    const int n0 = (tile & 7) * 64;

    CFrag c[4][4];
    #pragma unroll
    for (int i = 0; i < 4; ++i)
        #pragma unroll
        for (int j = 0; j < 4; ++j)
            #pragma unroll
            for (int e = 0; e < 8; ++e) c[i][j].f[e] = 0.f;

    for (int k0 = 0; k0 < DMODEL; k0 += 32) {
        v16bf a[4];
        #pragma unroll
        for (int i = 0; i < 4; ++i)
            a[i] = load_frag_a(xb + (size_t)(m0 + 16 * i) * DMODEL, DMODEL, k0, lane);
        #pragma unroll
        for (int j = 0; j < 4; ++j) {
            v16bf b = load_frag_b(wt + (size_t)(n0 + 16 * j) * DMODEL, DMODEL, k0, lane);
            #pragma unroll
            for (int i = 0; i < 4; ++i)
                c[i][j].v = wmma_bf16(a[i], b, c[i][j].v);
        }
    }

    const int hi = (lane >> 4) & 1;
    const int ln = lane & 15;
    #pragma unroll
    for (int i = 0; i < 4; ++i) {
        #pragma unroll
        for (int j = 0; j < 4; ++j) {
            const int col = n0 + 16 * j + ln;
            const int h = col >> 6, d = col & (DK - 1);
            const float bias_n = bias[col];
            #pragma unroll
            for (int v = 0; v < 8; ++v) {
                const int row = m0 + 16 * i + v + 8 * hi;     // 0..BL-1
                const int b_  = row >> 12;                    // / SEQ
                const int l_  = row & (SEQ - 1);
                const int bh  = b_ * NHEADS + h;
                const bf16_t val = (bf16_t)(c[i][j].f[v] + bias_n);
                if (which == 2)      Vt[((size_t)bh * DK + d) * SEQ + l_] = val;
                else if (which == 1) Kh[((size_t)bh * SEQ + l_) * DK + d] = val;
                else                 Qh[((size_t)bh * SEQ + l_) * DK + d] = val;
            }
        }
    }
}

// ---------------------------------------------------------------------------
// Kernel 3: flash attention with physics bias.
//   Block = 256 threads = 8 waves, all on the SAME head, 128 query rows.
//   Per 64-key iteration the block cooperatively stages the K tile (64x64)
//   and V tile (64x64, d-major) into LDS via async global->LDS, so K/V are
//   read from L2 once per 128 query rows instead of once per 16 (8x less
//   L2 traffic).  Each wave: S = Q.K^T (WMMA from LDS K), bias + online
//   softmax, P through its private LDS slab, O += P.V (WMMA from LDS V).
// ---------------------------------------------------------------------------
__global__ void __launch_bounds__(256) pba_flash_attn_kernel(
    const bf16_t* __restrict__ Qh, const bf16_t* __restrict__ Kh, const bf16_t* __restrict__ Vt,
    bf16_t* __restrict__ ctx, const float* __restrict__ gamma_p)
{
    __shared__ bf16_t Klds[64][72];        // [key][d]    9216 B
    __shared__ bf16_t Vlds[64][72];        // [d][key]    9216 B
    __shared__ bf16_t Plds[8][16][72];     // per-wave P 18432 B

    const int lane = threadIdx.x & 31;
    const int wave = threadIdx.x >> 5;
    const int bh   = blockIdx.x >> 5;              // 32 query-blocks per head
    const int row0 = (blockIdx.x & 31) * 128 + wave * 16;
    const int hi = (lane >> 4) & 1;
    const int ln = lane & 15;

    const float gamma   = gamma_p[0];
    const float scale   = 0.125f;                  // 1/sqrt(dk)
    const float negbias = gamma * (1.0f / 64.0f);  // gamma / sqrt(L)
    const float log2e   = 1.4426950408889634f;

    const bf16_t* qbase = Qh + ((size_t)bh * SEQ + row0) * DK;
    const bf16_t* kbase = Kh + (size_t)bh * SEQ * DK;
    const bf16_t* vbase = Vt + (size_t)bh * DK * SEQ;

    const v16bf qa0 = load_frag_a(qbase, DK, 0, lane);
    const v16bf qa1 = load_frag_a(qbase, DK, 32, lane);

    CFrag o[4];
    float m_row[8], l_row[8];
    #pragma unroll
    for (int n = 0; n < 4; ++n)
        #pragma unroll
        for (int e = 0; e < 8; ++e) o[n].f[e] = 0.f;
    #pragma unroll
    for (int v = 0; v < 8; ++v) { m_row[v] = -3.0e38f; l_row[v] = 0.f; }

    for (int j0 = 0; j0 < SEQ; j0 += 64) {
        // ---- cooperative async stage of K and V tiles (16B chunks) ----
        #pragma unroll
        for (int cc = 0; cc < 2; ++cc) {
            const int c   = (int)threadIdx.x + cc * 256;   // 0..511
            const int row = c >> 3;                         // key (K) / d (V)
            const int c16 = c & 7;                          // 16B chunk in row
            pba_copy16_to_lds(kbase + (size_t)(j0 + row) * DK + c16 * 8, &Klds[row][c16 * 8]);
            pba_copy16_to_lds(vbase + (size_t)row * SEQ + j0 + c16 * 8, &Vlds[row][c16 * 8]);
        }
        pba_wait_async();
        __syncthreads();

        // ---- scores: 4 fragments of 16 keys each, K from LDS ----
        CFrag s[4];
        #pragma unroll
        for (int t = 0; t < 4; ++t) {
            CFrag acc;
            #pragma unroll
            for (int e = 0; e < 8; ++e) acc.f[e] = 0.f;
            acc.v = wmma_bf16(qa0, load_frag_b(&Klds[16 * t][0], 72, 0, lane), acc.v);
            acc.v = wmma_bf16(qa1, load_frag_b(&Klds[16 * t][0], 72, 32, lane), acc.v);
            s[t] = acc;
        }

        // ---- bias + online softmax per row ----
        float alpha[8];
        #pragma unroll
        for (int v = 0; v < 8; ++v) {
            const int i_idx = row0 + v + 8 * hi;
            float xv[4];
            float mx = -3.0e38f;
            #pragma unroll
            for (int t = 0; t < 4; ++t) {
                const int j_idx = j0 + 16 * t + ln;
                const float dist = fabsf((float)(i_idx - j_idx));
                xv[t] = s[t].f[v] * scale - negbias * sqrtf(dist);
                mx = fmaxf(mx, xv[t]);
            }
            #pragma unroll
            for (int msk = 1; msk <= 8; msk <<= 1) mx = fmaxf(mx, __shfl_xor(mx, msk, 32));
            const float m_new = fmaxf(m_row[v], mx);
            const float a_    = exp2f((m_row[v] - m_new) * log2e);
            float ps = 0.f;
            #pragma unroll
            for (int t = 0; t < 4; ++t) {
                const float p = exp2f((xv[t] - m_new) * log2e);
                ps += p;
                Plds[wave][v + 8 * hi][16 * t + ln] = (bf16_t)p;
            }
            #pragma unroll
            for (int msk = 1; msk <= 8; msk <<= 1) ps += __shfl_xor(ps, msk, 32);
            l_row[v] = l_row[v] * a_ + ps;
            m_row[v] = m_new;
            alpha[v] = a_;
        }
        #pragma unroll
        for (int n = 0; n < 4; ++n)
            #pragma unroll
            for (int v = 0; v < 8; ++v) o[n].f[v] *= alpha[v];

        // in-wave LDS store -> load handoff for P
        asm volatile("s_wait_dscnt 0" ::: "memory");

        // ---- O += P @ V (P and V from LDS) ----
        const v16bf pa0 = load_frag_a(&Plds[wave][0][0], 72, 0, lane);
        const v16bf pa1 = load_frag_a(&Plds[wave][0][0], 72, 32, lane);
        #pragma unroll
        for (int n = 0; n < 4; ++n) {
            o[n].v = wmma_bf16(pa0, load_frag_b(&Vlds[16 * n][0], 72, 0, lane), o[n].v);
            o[n].v = wmma_bf16(pa1, load_frag_b(&Vlds[16 * n][0], 72, 32, lane), o[n].v);
        }
        __syncthreads();   // protect K/V LDS tiles before next stage
    }

    // ---- normalize + store ctx (bf16, [b][l][h*dk+d]) ----
    const int b_ = bh >> 3, h_ = bh & (NHEADS - 1);
    #pragma unroll
    for (int v = 0; v < 8; ++v) {
        const float inv_l = 1.0f / l_row[v];
        const int l_ = row0 + v + 8 * hi;
        #pragma unroll
        for (int n = 0; n < 4; ++n) {
            const int d = 16 * n + ln;
            ctx[((size_t)(b_ * SEQ + l_)) * DMODEL + h_ * DK + d] = (bf16_t)(o[n].f[v] * inv_l);
        }
    }
}

// ---------------------------------------------------------------------------
// Kernel 4: output projection GEMM, fp32 result to d_out
// ---------------------------------------------------------------------------
__global__ void __launch_bounds__(128) pba_out_gemm_kernel(
    const bf16_t* __restrict__ ctx, const bf16_t* __restrict__ wto,
    const float* __restrict__ bo, float* __restrict__ out)
{
    const int lane = threadIdx.x & 31;
    const int wave = threadIdx.x >> 5;
    const int tile = blockIdx.x * 4 + wave;
    const int m0 = (tile >> 3) * 64;
    const int n0 = (tile & 7) * 64;

    CFrag c[4][4];
    #pragma unroll
    for (int i = 0; i < 4; ++i)
        #pragma unroll
        for (int j = 0; j < 4; ++j)
            #pragma unroll
            for (int e = 0; e < 8; ++e) c[i][j].f[e] = 0.f;

    for (int k0 = 0; k0 < DMODEL; k0 += 32) {
        v16bf a[4];
        #pragma unroll
        for (int i = 0; i < 4; ++i)
            a[i] = load_frag_a(ctx + (size_t)(m0 + 16 * i) * DMODEL, DMODEL, k0, lane);
        #pragma unroll
        for (int j = 0; j < 4; ++j) {
            v16bf b = load_frag_b(wto + (size_t)(n0 + 16 * j) * DMODEL, DMODEL, k0, lane);
            #pragma unroll
            for (int i = 0; i < 4; ++i)
                c[i][j].v = wmma_bf16(a[i], b, c[i][j].v);
        }
    }

    const int hi = (lane >> 4) & 1;
    const int ln = lane & 15;
    #pragma unroll
    for (int i = 0; i < 4; ++i) {
        #pragma unroll
        for (int j = 0; j < 4; ++j) {
            const int col = n0 + 16 * j + ln;
            const float bias_n = bo[col];
            #pragma unroll
            for (int v = 0; v < 8; ++v) {
                const int row = m0 + 16 * i + v + 8 * hi;
                out[(size_t)row * DMODEL + col] = c[i][j].f[v] + bias_n;
            }
        }
    }
}

// ---------------------------------------------------------------------------
// Host launcher
// ---------------------------------------------------------------------------
extern "C" void kernel_launch(void* const* d_in, const int* in_sizes, int n_in,
                              void* d_out, int out_size, void* d_ws, size_t ws_size,
                              hipStream_t stream) {
    const float* x     = (const float*)d_in[0];
    const float* Wq    = (const float*)d_in[1];
    const float* bq    = (const float*)d_in[2];
    const float* Wk    = (const float*)d_in[3];
    const float* bk    = (const float*)d_in[4];
    const float* Wv    = (const float*)d_in[5];
    const float* bv    = (const float*)d_in[6];
    const float* Wo    = (const float*)d_in[7];
    const float* bo    = (const float*)d_in[8];
    const float* gamma = (const float*)d_in[9];
    float* out = (float*)d_out;

    char* ws = (char*)d_ws;
    size_t off = 0;
    auto take = [&](size_t bytes) -> char* {
        char* p = ws + off;
        off = (off + bytes + 255) & ~(size_t)255;
        return p;
    };

    bf16_t* xb   = (bf16_t*)take((size_t)BL * DMODEL * sizeof(bf16_t));      // 8 MB
    bf16_t* wtq  = (bf16_t*)take((size_t)DMODEL * DMODEL * sizeof(bf16_t));  // 512 KB
    bf16_t* wtk  = (bf16_t*)take((size_t)DMODEL * DMODEL * sizeof(bf16_t));
    bf16_t* wtv  = (bf16_t*)take((size_t)DMODEL * DMODEL * sizeof(bf16_t));
    bf16_t* wto  = (bf16_t*)take((size_t)DMODEL * DMODEL * sizeof(bf16_t));
    bf16_t* Qh   = (bf16_t*)take((size_t)BH * SEQ * DK * sizeof(bf16_t));    // 8 MB
    bf16_t* Kh   = (bf16_t*)take((size_t)BH * SEQ * DK * sizeof(bf16_t));    // 8 MB
    bf16_t* Vt   = (bf16_t*)take((size_t)BH * DK * SEQ * sizeof(bf16_t));    // 8 MB (transposed)
    bf16_t* ctxb = (bf16_t*)take((size_t)BL * DMODEL * sizeof(bf16_t));      // 8 MB

    // 1) precision convert + weight transpose
    pba_convert_x_kernel<<<(BL * DMODEL) / 256, 256, 0, stream>>>(x, xb, BL * DMODEL);
    pba_transpose_w_kernel<<<dim3(DMODEL * DMODEL / 256, 1, 4), 256, 0, stream>>>(
        Wq, Wk, Wv, Wo, wtq, wtk, wtv, wto);

    // 2) QKV projections (z = 0,1,2)
    pba_qkv_gemm_kernel<<<dim3(256, 1, 3), 128, 0, stream>>>(
        xb, wtq, wtk, wtv, bq, bk, bv, Qh, Kh, Vt);

    // 3) flash attention with physics bias (8 waves/block share K/V tiles)
    pba_flash_attn_kernel<<<BH * 32, 256, 0, stream>>>(Qh, Kh, Vt, ctxb, gamma);

    // 4) output projection -> fp32 out
    pba_out_gemm_kernel<<<256, 128, 0, stream>>>(ctxb, wto, bo, out);
}
